// MDMF_CLIPFSAR_17789754540999
// MI455X (gfx1250) — matrix operand order repaired and checked by reading, not compiled, not run on hardware
//
#include <hip/hip_runtime.h>

// ---------------------------------------------------------------------------
// MI455X / gfx1250, wave32. WMMA bf16 16x16x32 GEMM pipeline with 64x64
// register macro-tiles (16 WMMAs per K-step, 1 b128 load per WMMA).
// ---------------------------------------------------------------------------

typedef __attribute__((ext_vector_type(16))) __bf16 v16bf;
typedef __attribute__((ext_vector_type(8)))  float  v8f;

#define D_MODEL 512
#define MLPD    2048
#define T_SEQ   8
#define NSUP    25
#define NQRY    1024
#define NSEQ    1049          // NSUP + NQRY sequences
#define R_REAL  8392          // NSEQ * T_SEQ
#define R_PAD   8448          // padded to multiple of 64 (132 row-tiles of 64)
#define SEQ_PAD (R_PAD / 8)   // 1056 pseudo-sequences incl. padding
#define SF_ROWS 200           // NSUP * T_SEQ
#define TF_ROWS 8192          // NQRY * T_SEQ
#define NCOLPAD 208           // SF_ROWS padded to multiple of 16 (dists stride)
#define COS_EPS_C 0.01f
#define LN_EPS_C  1e-5f
#define LBDA_C    0.5f

union FragU { uint4 u[2]; v16bf v; };

// A-fragment (16x32 bf16, MxK): lane half h holds K = {k0+8h .. k0+8h+7} in
// VGPR0..3 and K = {k0+16+8h ..} in VGPR4..7 (ISA 7.12.2 16-bit A layout).
__device__ inline v16bf load_fragA(const __bf16* __restrict__ p) {
  FragU f;
  f.u[0] = *(const uint4*)(p);        // 8 bf16: K base+0..7
  f.u[1] = *(const uint4*)(p + 16);   // 8 bf16: K base+16..23
  return f.v;
}
// B-fragment (32x16 bf16, KxN): lane half h holds K = {k0+8h .. k0+8h+7} in
// VGPR0..3 and K = {k0+16+8h ..} in VGPR4..7 (same K order as A).
__device__ inline v16bf load_fragB(const __bf16* __restrict__ p) {
  FragU f;
  f.u[0] = *(const uint4*)(p);        // K base+0..7
  f.u[1] = *(const uint4*)(p + 16);   // K base+16..23
  return f.v;
}

// ---------------------------------------------------------------------------
// WMMA GEMM, 64x64 C tile per wave (4x4 accumulators):
// C[M,N] = act( A[M,K](bf16) * Bt[N,K]^T(bf16) + bias + residual )
// M, (grid N*64) multiples of 64. Stores guarded by n < n_valid.
// ---------------------------------------------------------------------------
__global__ __launch_bounds__(32) void wmma_gemm_bf16_t4(
    const __bf16* __restrict__ A, const __bf16* __restrict__ Bt,
    int N, int K,
    const float* __restrict__ bias, const float* __restrict__ residual,
    int act_gelu, float* __restrict__ Cf, __bf16* __restrict__ Cb,
    int n_valid)
{
  const int lane = threadIdx.x;
  const int half = lane >> 4, idx = lane & 15;
  const int m0 = blockIdx.y * 64, n0 = blockIdx.x * 64;

  const __bf16* aP[4];
  const __bf16* bP[4];
#pragma unroll
  for (int t = 0; t < 4; ++t) {
    aP[t] = A  + (size_t)(m0 + t * 16 + idx) * K + half * 8;
    bP[t] = Bt + (size_t)(n0 + t * 16 + idx) * K + half * 8;
  }

  v8f acc[4][4];
#pragma unroll
  for (int i = 0; i < 4; ++i)
#pragma unroll
    for (int j = 0; j < 4; ++j)
      acc[i][j] = (v8f){0.f, 0.f, 0.f, 0.f, 0.f, 0.f, 0.f, 0.f};

  for (int k0 = 0; k0 < K; k0 += 32) {
    v16bf a[4], b[4];
#pragma unroll
    for (int t = 0; t < 4; ++t) a[t] = load_fragA(aP[t] + k0);
#pragma unroll
    for (int t = 0; t < 4; ++t) b[t] = load_fragB(bP[t] + k0);
#pragma unroll
    for (int mt = 0; mt < 4; ++mt)
#pragma unroll
      for (int nt = 0; nt < 4; ++nt)
        acc[mt][nt] = __builtin_amdgcn_wmma_f32_16x16x32_bf16(
            false, a[mt], false, b[nt], (short)0, acc[mt][nt], false, false);
  }

#pragma unroll
  for (int nt = 0; nt < 4; ++nt) {
    const int n = n0 + nt * 16 + idx;
    if (n >= n_valid) continue;
    const float bv = bias ? bias[n] : 0.f;
#pragma unroll
    for (int mt = 0; mt < 4; ++mt) {
#pragma unroll
      for (int r = 0; r < 8; ++r) {
        const int m = m0 + mt * 16 + r + 8 * half;  // C layout: M = r + 8*half
        float v = acc[mt][nt][r] + bv;
        if (residual) v += residual[(size_t)m * N + n];
        if (act_gelu) v = 0.5f * v * (1.f + erff(v * 0.70710678118654752f));
        if (Cf) Cf[(size_t)m * N + n] = v;
        if (Cb) Cb[(size_t)m * N + n] = (__bf16)v;
      }
    }
  }
}

// ---------------------------------------------------------------------------
// Cos-sim GEMM (64x64 tiles): dists[i,j] = 1 - (tf_i.sf_j)/(|tf_i||sf_j|+eps)
// tf rows = Yb[SF_ROWS + i], sf rows = Yb[j]. Stores guarded by j < NCOLPAD
// (B rows 208..255 read defined target rows; their norms are also defined).
// ---------------------------------------------------------------------------
__global__ __launch_bounds__(32) void wmma_cossim_t4(
    const __bf16* __restrict__ Yb, const float* __restrict__ norms,
    float* __restrict__ dists)
{
  const int lane = threadIdx.x;
  const int half = lane >> 4, idx = lane & 15;
  const int i0 = blockIdx.y * 64;   // tf tile base
  const int j0 = blockIdx.x * 64;   // sf tile base

  const __bf16* aP[4];
  const __bf16* bP[4];
#pragma unroll
  for (int t = 0; t < 4; ++t) {
    aP[t] = Yb + (size_t)(SF_ROWS + i0 + t * 16 + idx) * D_MODEL + half * 8;
    bP[t] = Yb + (size_t)(j0 + t * 16 + idx) * D_MODEL + half * 8;
  }

  v8f acc[4][4];
#pragma unroll
  for (int i = 0; i < 4; ++i)
#pragma unroll
    for (int j = 0; j < 4; ++j)
      acc[i][j] = (v8f){0.f, 0.f, 0.f, 0.f, 0.f, 0.f, 0.f, 0.f};

  for (int k0 = 0; k0 < D_MODEL; k0 += 32) {
    v16bf a[4], b[4];
#pragma unroll
    for (int t = 0; t < 4; ++t) a[t] = load_fragA(aP[t] + k0);
#pragma unroll
    for (int t = 0; t < 4; ++t) b[t] = load_fragB(bP[t] + k0);
#pragma unroll
    for (int mt = 0; mt < 4; ++mt)
#pragma unroll
      for (int nt = 0; nt < 4; ++nt)
        acc[mt][nt] = __builtin_amdgcn_wmma_f32_16x16x32_bf16(
            false, a[mt], false, b[nt], (short)0, acc[mt][nt], false, false);
  }

#pragma unroll
  for (int nt = 0; nt < 4; ++nt) {
    const int j = j0 + nt * 16 + idx;
    if (j >= NCOLPAD) continue;
    const float nS = norms[j];
#pragma unroll
    for (int mt = 0; mt < 4; ++mt) {
#pragma unroll
      for (int r = 0; r < 8; ++r) {
        const int i = i0 + mt * 16 + r + 8 * half;
        const float sim = acc[mt][nt][r] / (norms[SF_ROWS + i] * nS + COS_EPS_C);
        dists[(size_t)i * NCOLPAD + j] = 1.f - sim;
      }
    }
  }
}

// ---------------------------------------------------------------------------
// Weight transpose + f32->bf16: W[K][N] -> Wt[N][K]
// ---------------------------------------------------------------------------
__global__ void transpose_to_bf16(const float* __restrict__ W,
                                  __bf16* __restrict__ Wt, int K, int N)
{
  const int tid = blockIdx.x * blockDim.x + threadIdx.x;
  if (tid >= K * N) return;
  const int k = tid / N, n = tid % N;
  Wt[(size_t)n * K + k] = (__bf16)W[tid];
}

// ---------------------------------------------------------------------------
// Concat(support, target) + LayerNorm. Wave per row; pad rows get zeros.
// ---------------------------------------------------------------------------
__global__ __launch_bounds__(32) void ln_concat(
    const float* __restrict__ sup, const float* __restrict__ tgt,
    const float* __restrict__ g, const float* __restrict__ bta,
    float* __restrict__ X, __bf16* __restrict__ XN)
{
  const int r = blockIdx.x, lane = threadIdx.x;
  const float* src = nullptr;
  if (r < SF_ROWS)      src = sup + (size_t)r * D_MODEL;
  else if (r < R_REAL)  src = tgt + (size_t)(r - SF_ROWS) * D_MODEL;
  float vals[16];
  float s = 0.f;
#pragma unroll
  for (int i = 0; i < 16; ++i) {
    vals[i] = src ? src[i * 32 + lane] : 0.f;
    s += vals[i];
  }
  for (int o = 16; o > 0; o >>= 1) s += __shfl_xor(s, o, 32);
  const float mu = s * (1.f / D_MODEL);
  float q = 0.f;
#pragma unroll
  for (int i = 0; i < 16; ++i) { const float d = vals[i] - mu; q += d * d; }
  for (int o = 16; o > 0; o >>= 1) q += __shfl_xor(q, o, 32);
  const float rstd = rsqrtf(q * (1.f / D_MODEL) + LN_EPS_C);
#pragma unroll
  for (int i = 0; i < 16; ++i) {
    const int c = i * 32 + lane;
    X[(size_t)r * D_MODEL + c] = vals[i];
    const float nv = src ? ((vals[i] - mu) * rstd * g[c] + bta[c]) : 0.f;
    XN[(size_t)r * D_MODEL + c] = (__bf16)nv;
  }
}

// ---------------------------------------------------------------------------
// Tiny 8x8-per-head attention, one thread per (seq, head, query-row).
// Pseudo-sequences >= NSEQ are padding -> zero output rows.
// ---------------------------------------------------------------------------
__global__ void attention_kernel(const __bf16* __restrict__ Q,
                                 const __bf16* __restrict__ Km,
                                 const __bf16* __restrict__ V,
                                 __bf16* __restrict__ O)
{
  const int tid = blockIdx.x * blockDim.x + threadIdx.x;
  if (tid >= SEQ_PAD * 64) return;
  const int i = tid & 7, h = (tid >> 3) & 7, seq = tid >> 6;
  const size_t rowO = (size_t)(seq * T_SEQ + i) * D_MODEL + h * 64;
  if (seq >= NSEQ) {
    for (int d = 0; d < 64; ++d) O[rowO + d] = (__bf16)0.f;
    return;
  }
  float att[8];
  for (int j = 0; j < 8; ++j) {
    const size_t rowK = (size_t)(seq * T_SEQ + j) * D_MODEL + h * 64;
    float sdot = 0.f;
    for (int d = 0; d < 64; ++d) sdot += (float)Q[rowO + d] * (float)Km[rowK + d];
    att[j] = sdot * 0.125f;                       // DH^-0.5 = 1/8
  }
  float mx = att[0];
  for (int j = 1; j < 8; ++j) mx = fmaxf(mx, att[j]);
  float den = 0.f;
  for (int j = 0; j < 8; ++j) { att[j] = __expf(att[j] - mx); den += att[j]; }
  const float inv = 1.f / den;
  for (int j = 0; j < 8; ++j) att[j] *= inv;
  for (int d = 0; d < 64; ++d) {
    float o = 0.f;
    for (int j = 0; j < 8; ++j)
      o += att[j] * (float)V[(size_t)(seq * T_SEQ + j) * D_MODEL + h * 64 + d];
    O[rowO + d] = (__bf16)o;
  }
}

// ---------------------------------------------------------------------------
// Row L2 norms of fp32 Y. Wave per row.
// ---------------------------------------------------------------------------
__global__ __launch_bounds__(32) void rownorm_kernel(const float* __restrict__ Y,
                                                     float* __restrict__ norms)
{
  const int r = blockIdx.x, lane = threadIdx.x;
  float s = 0.f;
#pragma unroll
  for (int i = 0; i < 16; ++i) {
    const float v = Y[(size_t)r * D_MODEL + i * 32 + lane];
    s += v * v;
  }
  for (int o = 16; o > 0; o >>= 1) s += __shfl_xor(s, o, 32);
  if (lane == 0) norms[r] = sqrtf(s);
}

// ---------------------------------------------------------------------------
// OTAM soft-DTW cumulative distance (one direction). dist(l,m)=base[l*s0+m*s1],
// 8x8 grid padded with a zero column on each side (M=10). Vertical term (l-1,m) only
// included at m==1 and m==9, matching the reference's `incl` mask.
// ---------------------------------------------------------------------------
__device__ float otam_dir(const float* __restrict__ base, int s0, int s1)
{
  float prev[10], cur[10];
  float c = 0.f;
  prev[0] = 0.f;
  for (int m = 1; m <= 9; ++m) {
    const float dm = (m <= 8) ? base[(m - 1) * s1] : 0.f;
    c += dm;
    prev[m] = c;
  }
  const float invl = 1.f / LBDA_C;
  for (int l = 1; l < T_SEQ; ++l) {
    cur[0] = 0.f;
    for (int m = 1; m <= 9; ++m) {
      const float dm = (m <= 8) ? base[l * s0 + (m - 1) * s1] : 0.f;
      const float a = -prev[m - 1] * invl;
      const float b = -cur[m - 1] * invl;
      const bool uc = (m == 1) || (m == 9);
      const float cc = -prev[m] * invl;
      float mx = fmaxf(a, b);
      if (uc) mx = fmaxf(mx, cc);
      const float sm = __expf(a - mx) + __expf(b - mx) + (uc ? __expf(cc - mx) : 0.f);
      cur[m] = dm - LBDA_C * (mx + __logf(sm));
    }
    for (int m = 0; m <= 9; ++m) prev[m] = cur[m];
  }
  return prev[9];
}

__global__ void otam_kernel(const float* __restrict__ dists, float* __restrict__ out)
{
  const int tid = blockIdx.x * blockDim.x + threadIdx.x;
  if (tid >= NQRY * NSUP) return;
  const int s = tid % NSUP, q = tid / NSUP;
  const float* base = dists + (size_t)(q * T_SEQ) * NCOLPAD + s * T_SEQ;
  const float c1 = otam_dir(base, NCOLPAD, 1);   // [tq, ts]
  const float c2 = otam_dir(base, 1, NCOLPAD);   // transposed
  out[tid] = -(c1 + c2);
}

// ---------------------------------------------------------------------------
// Host-side orchestration
// ---------------------------------------------------------------------------
extern "C" void kernel_launch(void* const* d_in, const int* in_sizes, int n_in,
                              void* d_out, int out_size, void* d_ws, size_t ws_size,
                              hipStream_t stream)
{
  (void)in_sizes; (void)n_in; (void)out_size; (void)ws_size;
  const float* sup  = (const float*)d_in[0];
  const float* tgt  = (const float*)d_in[1];
  const float* ln_g = (const float*)d_in[2];
  const float* ln_b = (const float*)d_in[3];
  const float* Wq   = (const float*)d_in[4];
  const float* Wk   = (const float*)d_in[5];
  const float* Wv   = (const float*)d_in[6];
  const float* Wo   = (const float*)d_in[7];
  const float* bo   = (const float*)d_in[8];
  const float* W1   = (const float*)d_in[9];
  const float* b1   = (const float*)d_in[10];
  const float* W2   = (const float*)d_in[11];
  const float* b2   = (const float*)d_in[12];
  float* out = (float*)d_out;
  char* ws = (char*)d_ws;

  // ---- workspace layout (bytes, all 256-aligned); aggressive aliasing ----
  const size_t o_wqt = 0;
  const size_t o_wkt = o_wqt + (size_t)D_MODEL * D_MODEL * 2;
  const size_t o_wvt = o_wkt + (size_t)D_MODEL * D_MODEL * 2;
  const size_t o_wot = o_wvt + (size_t)D_MODEL * D_MODEL * 2;
  const size_t o_w1t = o_wot + (size_t)D_MODEL * D_MODEL * 2;
  const size_t o_w2t = o_w1t + (size_t)D_MODEL * MLPD * 2;
  const size_t o_X   = o_w2t + (size_t)D_MODEL * MLPD * 2;          // fp32 [R_PAD,512]
  const size_t o_XN  = o_X   + (size_t)R_PAD * D_MODEL * 4;         // bf16 [R_PAD,512]
  const size_t o_QKV = o_XN  + (size_t)R_PAD * D_MODEL * 2;         // 3x bf16
  const size_t o_O   = o_QKV + 3 * (size_t)R_PAD * D_MODEL * 2;     // bf16
  const size_t o_X1  = o_O   + (size_t)R_PAD * D_MODEL * 2;         // fp32
  const size_t o_Yb  = o_X1  + (size_t)R_PAD * D_MODEL * 4;         // bf16
  const size_t o_nrm = o_Yb  + (size_t)R_PAD * D_MODEL * 2;         // fp32 [R_PAD]
  const size_t o_dst = (o_nrm + (size_t)R_PAD * 4 + 255) & ~(size_t)255;
  // Aliases (lifetimes verified against the launch order below):
  const size_t o_X1b = o_XN;    // XN dead after QKV GEMMs
  const size_t o_H   = o_QKV;   // QKV+O region == R_PAD*MLPD*2 bytes exactly
  const size_t o_Y   = o_X;     // X dead after attention-residual GEMM

  __bf16* wqt = (__bf16*)(ws + o_wqt);
  __bf16* wkt = (__bf16*)(ws + o_wkt);
  __bf16* wvt = (__bf16*)(ws + o_wvt);
  __bf16* wot = (__bf16*)(ws + o_wot);
  __bf16* w1t = (__bf16*)(ws + o_w1t);
  __bf16* w2t = (__bf16*)(ws + o_w2t);
  float*  X   = (float*) (ws + o_X);
  __bf16* XN  = (__bf16*)(ws + o_XN);
  __bf16* Qb  = (__bf16*)(ws + o_QKV);
  __bf16* Kb  = Qb + (size_t)R_PAD * D_MODEL;
  __bf16* Vb  = Kb + (size_t)R_PAD * D_MODEL;
  __bf16* Ob  = (__bf16*)(ws + o_O);
  float*  X1  = (float*) (ws + o_X1);
  __bf16* X1b = (__bf16*)(ws + o_X1b);
  __bf16* Hb  = (__bf16*)(ws + o_H);
  float*  Yf  = (float*) (ws + o_Y);
  __bf16* Yb  = (__bf16*)(ws + o_Yb);
  float*  nrm = (float*) (ws + o_nrm);
  float*  dst = (float*) (ws + o_dst);

  // 1) weights -> bf16, transposed to [N][K]
  {
    const int nsq = D_MODEL * D_MODEL, nmlp = D_MODEL * MLPD;
    transpose_to_bf16<<<(nsq  + 255) / 256, 256, 0, stream>>>(Wq, wqt, D_MODEL, D_MODEL);
    transpose_to_bf16<<<(nsq  + 255) / 256, 256, 0, stream>>>(Wk, wkt, D_MODEL, D_MODEL);
    transpose_to_bf16<<<(nsq  + 255) / 256, 256, 0, stream>>>(Wv, wvt, D_MODEL, D_MODEL);
    transpose_to_bf16<<<(nsq  + 255) / 256, 256, 0, stream>>>(Wo, wot, D_MODEL, D_MODEL);
    transpose_to_bf16<<<(nmlp + 255) / 256, 256, 0, stream>>>(W1, w1t, D_MODEL, MLPD);
    transpose_to_bf16<<<(nmlp + 255) / 256, 256, 0, stream>>>(W2, w2t, MLPD, D_MODEL);
  }

  // 2) concat + layernorm
  ln_concat<<<R_PAD, 32, 0, stream>>>(sup, tgt, ln_g, ln_b, X, XN);

  // 3) QKV projections (WMMA, 64x64 tiles)
  dim3 gD(D_MODEL / 64, R_PAD / 64);
  wmma_gemm_bf16_t4<<<gD, 32, 0, stream>>>(XN, wqt, D_MODEL, D_MODEL,
                                           nullptr, nullptr, 0, nullptr, Qb, D_MODEL);
  wmma_gemm_bf16_t4<<<gD, 32, 0, stream>>>(XN, wkt, D_MODEL, D_MODEL,
                                           nullptr, nullptr, 0, nullptr, Kb, D_MODEL);
  wmma_gemm_bf16_t4<<<gD, 32, 0, stream>>>(XN, wvt, D_MODEL, D_MODEL,
                                           nullptr, nullptr, 0, nullptr, Vb, D_MODEL);

  // 4) attention (tiny, VALU)
  attention_kernel<<<(SEQ_PAD * 64 + 255) / 256, 256, 0, stream>>>(Qb, Kb, Vb, Ob);

  // 5) output projection + bias + residual -> X1 (fp32 + bf16)
  wmma_gemm_bf16_t4<<<gD, 32, 0, stream>>>(Ob, wot, D_MODEL, D_MODEL,
                                           bo, X, 0, X1, X1b, D_MODEL);

  // 6) MLP up-proj + exact GELU -> H (bf16)
  dim3 gM(MLPD / 64, R_PAD / 64);
  wmma_gemm_bf16_t4<<<gM, 32, 0, stream>>>(X1b, w1t, MLPD, D_MODEL,
                                           b1, nullptr, 1, nullptr, Hb, MLPD);

  // 7) MLP down-proj + bias + residual -> Y (fp32 + bf16)
  wmma_gemm_bf16_t4<<<gD, 32, 0, stream>>>(Hb, w2t, D_MODEL, MLPD,
                                           b2, X1, 0, Yf, Yb, D_MODEL);

  // 8) row norms (fp32)
  rownorm_kernel<<<R_REAL, 32, 0, stream>>>(Yf, nrm);

  // 9) cos-sim -> dists[TF_ROWS][NCOLPAD] (WMMA, 64x64 tiles, 4 col groups)
  dim3 gC(256 / 64, TF_ROWS / 64);
  wmma_cossim_t4<<<gC, 32, 0, stream>>>(Yb, nrm, dst);

  // 10) OTAM both directions -> logits [NQ, NS]
  otam_kernel<<<(NQRY * NSUP + 255) / 256, 256, 0, stream>>>(dst, out);
}
